// WorldModel_45054206935482
// MI455X (gfx1250) — matrix operand order repaired and checked by reading, not compiled, not run on hardware
//
#include <hip/hip_runtime.h>
#include <hip/hip_bf16.h>
#include <math.h>

typedef _Float16 f16;
typedef __attribute__((ext_vector_type(16))) _Float16 v16h;
typedef __attribute__((ext_vector_type(8)))  _Float16 v8h;
typedef __attribute__((ext_vector_type(8)))  float    v8f;

// Model constants
#define WB   32
#define WT   64
#define RECN 1024
#define LATN 1024
#define ACTD 6

__device__ __forceinline__ float silu_f(float x) { return x / (1.f + __expf(-x)); }

// ---- staging helpers: 16 contiguous K elements -> LDS (16B-aligned dst) ----
// Fast path: one/two wide loads. Slow path: clamped-address loads (no exec-branch
// serialization: address is selected, load always executes, value masked after).
__device__ __forceinline__ void stage16(const float* __restrict__ src, bool fast, bool rowOk,
                                        long base, int kbase, int K, f16* dst) {
  if (fast) {
    const float4* s = (const float4*)(src + base + kbase);
    float4 a0 = s[0], a1 = s[1], a2 = s[2], a3 = s[3];
    union { v8h v; f16 e[8]; } u0, u1;
    u0.e[0]=(f16)a0.x; u0.e[1]=(f16)a0.y; u0.e[2]=(f16)a0.z; u0.e[3]=(f16)a0.w;
    u0.e[4]=(f16)a1.x; u0.e[5]=(f16)a1.y; u0.e[6]=(f16)a1.z; u0.e[7]=(f16)a1.w;
    u1.e[0]=(f16)a2.x; u1.e[1]=(f16)a2.y; u1.e[2]=(f16)a2.z; u1.e[3]=(f16)a2.w;
    u1.e[4]=(f16)a3.x; u1.e[5]=(f16)a3.y; u1.e[6]=(f16)a3.z; u1.e[7]=(f16)a3.w;
    ((v8h*)dst)[0] = u0.v; ((v8h*)dst)[1] = u1.v;
  } else {
    union { v8h v; f16 e[8]; } u0, u1;
#pragma unroll
    for (int j = 0; j < 16; ++j) {
      int k = kbase + j;
      bool ok = rowOk && k < K;
      float v = src[ok ? base + k : 0];
      f16 h = ok ? (f16)v : (f16)0.f;
      if (j < 8) u0.e[j] = h; else u1.e[j - 8] = h;
    }
    ((v8h*)dst)[0] = u0.v; ((v8h*)dst)[1] = u1.v;
  }
}
__device__ __forceinline__ void stage16(const f16* __restrict__ src, bool fast, bool rowOk,
                                        long base, int kbase, int K, f16* dst) {
  if (fast) {
    const v8h* s = (const v8h*)(src + base + kbase);
    ((v8h*)dst)[0] = s[0]; ((v8h*)dst)[1] = s[1];
  } else {
    union { v8h v; f16 e[8]; } u0, u1;
#pragma unroll
    for (int j = 0; j < 16; ++j) {
      int k = kbase + j;
      bool ok = rowOk && k < K;
      f16 v = src[ok ? base + k : 0];
      f16 h = ok ? v : (f16)0.f;
      if (j < 8) u0.e[j] = h; else u1.e[j - 8] = h;
    }
    ((v8h*)dst)[0] = u0.v; ((v8h*)dst)[1] = u1.v;
  }
}

// ---------------- conversion kernels ----------------
__global__ void f2h_kernel(const float* __restrict__ s, f16* __restrict__ d, long n) {
  long i = (long)blockIdx.x * blockDim.x + threadIdx.x;
  if (i < n) d[i] = (f16)s[i];
}

// src (Z,R,C) f32 row-major -> dst (Z,C,ldOut) f16 (row stride ldOut >= R, for alignment pad)
__global__ void tr_f2h_kernel(const float* __restrict__ src, f16* __restrict__ dst,
                              int R, int Cc, int ldOut) {
  long z = blockIdx.z;
  const float* s = src + z * (long)R * Cc;
  f16* d = dst + z * (long)Cc * ldOut;
  int idx = blockIdx.x * blockDim.x + threadIdx.x;
  if (idx >= R * Cc) return;
  int r = idx / Cc, c = idx - r * Cc;
  d[(long)c * ldOut + r] = (f16)s[idx];
}

__global__ void zero_kernel(float* p, int n) {
  int i = blockIdx.x * blockDim.x + threadIdx.x;
  if (i < n) p[i] = 0.f;
}

// ---------------- WMMA GEMM (large-M): 64x64 block tile, 4 waves -----------------
template<typename AT>
__global__ __launch_bounds__(128) void gemm_wmma(
    const AT* __restrict__ A, int lda, long aZ,
    const f16* __restrict__ Bt, int ldb, long bZ,
    const float* __restrict__ bias,
    float* __restrict__ C, int ldc, long cZ,
    int M, int N, int K, int act)
{
  __shared__ __align__(16) f16 As[64 * 40];
  __shared__ __align__(16) f16 Bs[64 * 40];
  const AT*  Ap = A  + aZ * blockIdx.z;
  const f16* Bp = Bt + bZ * blockIdx.z;
  float*     Cp = C  + cZ * blockIdx.z;
  const int tid  = threadIdx.x;
  const int wave = tid >> 5, lane = tid & 31;
  const int bM = blockIdx.y * 64, bN = blockIdx.x * 64;
  const int r = tid >> 1, seg = (tid & 1) << 4;
  const int mA = bM + r, nB = bN + r;
  const bool aOk = mA < M, bOk = nB < N;
  const long aBase = aOk ? (long)mA * lda : 0;
  const long bBase = bOk ? (long)nB * ldb : 0;
  const int Kpad = (K + 31) & ~31;
  v8f acc[4] = {};
  const int mRow = (wave << 4) + (lane & 15);
  const int kA = (lane >> 4) << 3;   // A frag: K base 0 or 8 (ISA 7.12.2 16-bit A layout)
  const int kB = (lane >> 4) << 3;   // B frag: K base 0 or 8 (same layout as A)
  for (int k0 = 0; k0 < Kpad; k0 += 32) {
    stage16(Ap, aOk && (k0 + seg + 16 <= K), aOk, aBase, k0 + seg, K, &As[r * 40 + seg]);
    stage16(Bp, bOk && (k0 + seg + 16 <= K), bOk, bBase, k0 + seg, K, &Bs[r * 40 + seg]);
    __syncthreads();
    if (k0 + 32 < Kpad && bOk)
      __builtin_prefetch(&Bp[bBase + k0 + 32], 0, 1);   // global_prefetch_b8
    union { v16h v; v8h h[2]; } a, b;
    a.h[0] = *(const v8h*)&As[mRow * 40 + kA];
    a.h[1] = *(const v8h*)&As[mRow * 40 + kA + 16];
#pragma unroll
    for (int nt = 0; nt < 4; ++nt) {
      const int nRow = (nt << 4) + (lane & 15);
      b.h[0] = *(const v8h*)&Bs[nRow * 40 + kB];
      b.h[1] = *(const v8h*)&Bs[nRow * 40 + kB + 16];
      acc[nt] = __builtin_amdgcn_wmma_f32_16x16x32_f16(false, a.v, false, b.v,
                                                       (short)0, acc[nt], false, false);
    }
    __syncthreads();
  }
  const int mBase = bM + (wave << 4) + ((lane >> 4) << 3);
#pragma unroll
  for (int nt = 0; nt < 4; ++nt) {
    int n = bN + (nt << 4) + (lane & 15);
    if (n >= N) continue;
    float bv = bias ? bias[n] : 0.f;
#pragma unroll
    for (int j = 0; j < 8; ++j) {
      int m = mBase + j;
      if (m < M) {
        float v = acc[nt][j] + bv;
        if (act) v = silu_f(v);
        Cp[(long)m * ldc + n] = v;
      }
    }
  }
}

// ---------------- WMMA GEMM (M<=32, scan critical path): 32x128 block tile ----------------
// wave w -> m-tile (w&1), n-half (w>>1): all 4 waves productive for M=32.
__global__ __launch_bounds__(128) void gemm_wmma_m32(
    const float* __restrict__ A, int lda, long aZ,
    const f16* __restrict__ Bt, int ldb, long bZ,
    const float* __restrict__ bias,
    float* __restrict__ C, int ldc, long cZ,
    int M, int N, int K, int act)
{
  __shared__ __align__(16) f16 As[32 * 40];
  __shared__ __align__(16) f16 Bs[128 * 40];
  const float* Ap = A  + aZ * blockIdx.z;
  const f16*   Bp = Bt + bZ * blockIdx.z;
  float*       Cp = C  + cZ * blockIdx.z;
  const int tid  = threadIdx.x;
  const int wave = tid >> 5, lane = tid & 31;
  const int bM = blockIdx.y * 32, bN = blockIdx.x * 128;
  const int mA = bM + tid;          // threads 0..31 each stage one A row (32 halves)
  const int nB = bN + tid;          // every thread stages one B row (32 halves)
  const bool aOk = (tid < 32) && (mA < M), bOk = nB < N;
  const long aBase = aOk ? (long)mA * lda : 0;
  const long bBase = bOk ? (long)nB * ldb : 0;
  const int Kpad = (K + 31) & ~31;
  v8f acc[4] = {};
  const int mRow  = ((wave & 1) << 4) + (lane & 15);
  const int nBase = (wave >> 1) << 6;
  const int kA = (lane >> 4) << 3;
  const int kB = (lane >> 4) << 3;
  for (int k0 = 0; k0 < Kpad; k0 += 32) {
    if (tid < 32) {
      stage16(Ap, aOk && (k0 + 16 <= K), aOk, aBase, k0,      K, &As[tid * 40]);
      stage16(Ap, aOk && (k0 + 32 <= K), aOk, aBase, k0 + 16, K, &As[tid * 40 + 16]);
    }
    stage16(Bp, bOk && (k0 + 16 <= K), bOk, bBase, k0,      K, &Bs[tid * 40]);
    stage16(Bp, bOk && (k0 + 32 <= K), bOk, bBase, k0 + 16, K, &Bs[tid * 40 + 16]);
    __syncthreads();
    if (k0 + 32 < Kpad && bOk)
      __builtin_prefetch(&Bp[bBase + k0 + 32], 0, 1);
    union { v16h v; v8h h[2]; } a, b;
    a.h[0] = *(const v8h*)&As[mRow * 40 + kA];
    a.h[1] = *(const v8h*)&As[mRow * 40 + kA + 16];
#pragma unroll
    for (int nt = 0; nt < 4; ++nt) {
      const int nRow = nBase + (nt << 4) + (lane & 15);
      b.h[0] = *(const v8h*)&Bs[nRow * 40 + kB];
      b.h[1] = *(const v8h*)&Bs[nRow * 40 + kB + 16];
      acc[nt] = __builtin_amdgcn_wmma_f32_16x16x32_f16(false, a.v, false, b.v,
                                                       (short)0, acc[nt], false, false);
    }
    __syncthreads();
  }
  const int mBase = bM + ((wave & 1) << 4) + ((lane >> 4) << 3);
#pragma unroll
  for (int nt = 0; nt < 4; ++nt) {
    int n = bN + nBase + (nt << 4) + (lane & 15);
    if (n >= N) continue;
    float bv = bias ? bias[n] : 0.f;
#pragma unroll
    for (int j = 0; j < 8; ++j) {
      int m = mBase + j;
      if (m < M) {
        float v = acc[nt][j] + bv;
        if (act) v = silu_f(v);
        Cp[(long)m * ldc + n] = v;
      }
    }
  }
}

// ---------------- WMMA implicit-GEMM conv (stride 2, 4x4 kernel, pad 1) + SiLU -> f16 NCHW ----
template<typename AT>
__global__ __launch_bounds__(128) void conv_wmma(
    const AT* __restrict__ X, const f16* __restrict__ Wt, const float* __restrict__ bias,
    f16* __restrict__ Y, int NIMG, int IC, int IH, int IW, int OC, int OH, int OW)
{
  __shared__ __align__(16) f16 As[64 * 40];
  __shared__ __align__(16) f16 Bs[64 * 40];
  const int K = IC << 4;            // ic*4*4, same ordering as OIHW weights
  const int OHW = OH * OW;
  const int M = NIMG * OHW;
  const int tid  = threadIdx.x;
  const int wave = tid >> 5, lane = tid & 31;
  const int bM = blockIdx.y * 64, bN = blockIdx.x * 64;
  const int r = tid >> 1, seg = (tid & 1) << 4;
  const int mA = bM + r, nB = bN + r;
  const bool aOk = mA < M, bOk = nB < OC;
  const long bBase = bOk ? (long)nB * K : 0;
  const int Kpad = (K + 31) & ~31;
  int ib = 0, oy = 0, ox = 0;
  if (aOk) { ib = mA / OHW; int rem = mA - ib * OHW; oy = rem / OW; ox = rem - oy * OW; }
  v8f acc[4] = {};
  const int mRow = (wave << 4) + (lane & 15);
  const int kA = (lane >> 4) << 3;
  const int kB = (lane >> 4) << 3;
  for (int k0 = 0; k0 < Kpad; k0 += 32) {
    {
      // im2col gather: clamped-address loads, packed LDS store
      union { v8h v; f16 e[8]; } u0, u1;
#pragma unroll
      for (int j = 0; j < 16; ++j) {
        int k = k0 + seg + j;
        int ic = k >> 4, kh = (k >> 2) & 3, kw = k & 3;
        int iy = (oy << 1) - 1 + kh, ix = (ox << 1) - 1 + kw;
        bool ok = aOk && k < K && iy >= 0 && iy < IH && ix >= 0 && ix < IW;
        long idx = ok ? (((long)ib * IC + ic) * IH + iy) * IW + ix : 0;
        float v = (float)X[idx];
        f16 h = ok ? (f16)v : (f16)0.f;
        if (j < 8) u0.e[j] = h; else u1.e[j - 8] = h;
      }
      v8h* d = (v8h*)&As[r * 40 + seg];
      d[0] = u0.v; d[1] = u1.v;
    }
    stage16(Wt, bOk && (k0 + seg + 16 <= K), bOk, bBase, k0 + seg, K, &Bs[r * 40 + seg]);
    __syncthreads();
    if (k0 + 32 < Kpad && bOk)
      __builtin_prefetch(&Wt[bBase + k0 + 32], 0, 1);
    union { v16h v; v8h h[2]; } a, b;
    a.h[0] = *(const v8h*)&As[mRow * 40 + kA];
    a.h[1] = *(const v8h*)&As[mRow * 40 + kA + 16];
#pragma unroll
    for (int nt = 0; nt < 4; ++nt) {
      const int nRow = (nt << 4) + (lane & 15);
      b.h[0] = *(const v8h*)&Bs[nRow * 40 + kB];
      b.h[1] = *(const v8h*)&Bs[nRow * 40 + kB + 16];
      acc[nt] = __builtin_amdgcn_wmma_f32_16x16x32_f16(false, a.v, false, b.v,
                                                       (short)0, acc[nt], false, false);
    }
    __syncthreads();
  }
  const int mBase = bM + (wave << 4) + ((lane >> 4) << 3);
#pragma unroll
  for (int nt = 0; nt < 4; ++nt) {
    int n = bN + (nt << 4) + (lane & 15);
    if (n >= OC) continue;
    float bv = bias[n];
#pragma unroll
    for (int j = 0; j < 8; ++j) {
      int m = mBase + j;
      if (m < M) {
        int b2 = m / OHW; int rem = m - b2 * OHW;
        int oy2 = rem / OW, ox2 = rem - oy2 * OW;
        float v = silu_f(acc[nt][j] + bv);
        Y[(((long)b2 * OC + n) * OH + oy2) * OW + ox2] = (f16)v;
      }
    }
  }
}

// ---------------- scan glue kernels ----------------
__global__ void pack_xpost(const float* __restrict__ h, const float* __restrict__ embed,
                           float* __restrict__ xp, int t) {
  int i = blockIdx.x * blockDim.x + threadIdx.x;        // 32*2048
  if (i >= WB * 2048) return;
  int b = i >> 11, c = i & 2047;
  xp[i] = (c < RECN) ? h[b * RECN + c]
                     : embed[((long)(b * WT + t)) * LATN + (c - RECN)];
}

__global__ void pack_xfull(const float* __restrict__ h, const float* __restrict__ smp,
                           const float* __restrict__ actions, float* __restrict__ xf, int t) {
  int i = blockIdx.x * blockDim.x + threadIdx.x;        // 32*2080 (K padded)
  if (i >= WB * 2080) return;
  int b = i / 2080, c = i - b * 2080;
  float v = 0.f;
  if      (c < RECN)        v = h[b * RECN + c];
  else if (c < RECN + LATN) v = smp[b * LATN + (c - RECN)];
  else if (c < RECN + LATN + ACTD) v = actions[((long)(b * WT + t)) * ACTD + (c - RECN - LATN)];
  xf[i] = v;
}

__device__ __forceinline__ unsigned hash_u32(unsigned x) {
  x ^= x >> 16; x *= 0x7feb352dU; x ^= x >> 15; x *= 0x846ca68bU; x ^= x >> 16; return x;
}

// One wave per (batch, group): log_softmax for post & prior, Gumbel-max sample,
// write [h | sample | h | post_lp | prior_lp] slice of d_out for step t.
__global__ __launch_bounds__(128) void softmax_sample(
    const float* __restrict__ plog, const float* __restrict__ rlog, const float* __restrict__ h,
    float* __restrict__ smp, float* __restrict__ out, int t)
{
  int wave = threadIdx.x >> 5, lane = threadIdx.x & 31;
  int pair = blockIdx.x * 4 + wave;                     // 0..1023
  int b = pair >> 5, g = pair & 31;
  int cc = g * 32 + lane;
  // ---- post ----
  float x = plog[b * LATN + cc];
  float mx = x;
  for (int o = 16; o; o >>= 1) mx = fmaxf(mx, __shfl_xor(mx, o, 32));
  float e = __expf(x - mx), s = e;
  for (int o = 16; o; o >>= 1) s += __shfl_xor(s, o, 32);
  float lp = x - mx - __logf(s);
  unsigned u = hash_u32(((unsigned)t * 1024u + (unsigned)pair) * 37u + (unsigned)lane + 0x9e3779b9u);
  float uf = ((float)(u >> 8) + 0.5f) * (1.f / 16777216.f);
  float key = lp - __logf(-__logf(uf));
  int arg = lane;
  for (int o = 16; o; o >>= 1) {
    float ok = __shfl_xor(key, o, 32);
    int   oa = __shfl_xor(arg, o, 32);
    if (ok > key || (ok == key && oa < arg)) { key = ok; arg = oa; }
  }
  float onehot = (lane == arg) ? 1.f : 0.f;
  long ob = (long)b * WT * 5120 + (long)t * 5120;
  float hv = h[b * RECN + cc];
  out[ob + cc]        = hv;        // full_state[:1024] = h
  out[ob + 1024 + cc] = onehot;    // full_state[1024:] = sample (one-hot fwd value)
  out[ob + 2048 + cc] = hv;        // rs = h
  out[ob + 3072 + cc] = lp;        // post log-probs
  smp[b * LATN + cc]  = onehot;
  // ---- prior ----
  float y = rlog[b * LATN + cc];
  float my = y;
  for (int o = 16; o; o >>= 1) my = fmaxf(my, __shfl_xor(my, o, 32));
  float ey = __expf(y - my), sy = ey;
  for (int o = 16; o; o >>= 1) sy += __shfl_xor(sy, o, 32);
  out[ob + 4096 + cc] = y - my - __logf(sy);
}

__global__ void gru_combine(const float* __restrict__ gx, const float* __restrict__ gh,
                            const float* __restrict__ h, float* __restrict__ hn) {
  int i = blockIdx.x * blockDim.x + threadIdx.x;        // 32*1024
  if (i >= WB * RECN) return;
  int b = i >> 10, c = i & 1023;
  int nb = c >> 7, j = c & 127;
  int base = b * 3072 + nb * 384;
  float r = 1.f / (1.f + __expf(-(gx[base + j]       + gh[base + j])));
  float z = 1.f / (1.f + __expf(-(gx[base + 128 + j] + gh[base + 128 + j])));
  float n = tanhf(gx[base + 256 + j] + r * gh[base + 256 + j]);
  hn[i] = (1.f - z) * n + z * h[i];
}

// ---------------- host launcher ----------------
extern "C" void kernel_launch(void* const* d_in, const int* in_sizes, int n_in,
                              void* d_out, int out_size, void* d_ws, size_t ws_size,
                              hipStream_t stream) {
  (void)in_sizes; (void)n_in; (void)out_size; (void)ws_size;
  const float* obs      = (const float*)d_in[0];
  const float* actions  = (const float*)d_in[1];
  const float* enc_w[4] = {(const float*)d_in[2], (const float*)d_in[4],
                           (const float*)d_in[6], (const float*)d_in[8]};
  const float* enc_b[4] = {(const float*)d_in[3], (const float*)d_in[5],
                           (const float*)d_in[7], (const float*)d_in[9]};
  const float* enc_wl  = (const float*)d_in[10]; const float* enc_bl  = (const float*)d_in[11];
  const float* post_w1 = (const float*)d_in[12]; const float* post_b1 = (const float*)d_in[13];
  const float* post_w2 = (const float*)d_in[14]; const float* post_b2 = (const float*)d_in[15];
  const float* prior_w1= (const float*)d_in[16]; const float* prior_b1= (const float*)d_in[17];
  const float* prior_w2= (const float*)d_in[18]; const float* prior_b2= (const float*)d_in[19];
  const float* gru_wx  = (const float*)d_in[20]; const float* gru_bx  = (const float*)d_in[21];
  const float* gru_uh  = (const float*)d_in[22];
  float* out = (float*)d_out;

  char* ws = (char*)d_ws;
  size_t off = 0;
  auto alloc = [&](size_t bytes) -> void* {
    void* p = ws + off; off = (off + bytes + 255) & ~(size_t)255; return p;
  };
  const int CH = 512;                                   // images per encoder chunk
  f16* a1c  = (f16*)alloc((size_t)CH * 32 * 32 * 32 * 2);
  f16* a2c  = (f16*)alloc((size_t)CH * 64 * 16 * 16 * 2);
  f16* a3c  = (f16*)alloc((size_t)CH * 128 * 8 * 8 * 2);
  f16* a4c  = (f16*)alloc((size_t)CH * 4096 * 2);
  float* embed = (float*)alloc((size_t)2048 * 1024 * 4);
  f16* w1c  = (f16*)alloc(32  * 48   * 2);
  f16* w2c  = (f16*)alloc(64  * 512  * 2);
  f16* w3c  = (f16*)alloc(128 * 1024 * 2);
  f16* w4c  = (f16*)alloc(256 * 2048 * 2);
  f16* wlT  = (f16*)alloc((size_t)4096 * 1024 * 2);
  f16* pw1T = (f16*)alloc((size_t)2048 * 1024 * 2);
  f16* pw2T = (f16*)alloc((size_t)1024 * 1024 * 2);
  f16* rw1T = (f16*)alloc((size_t)1024 * 1024 * 2);
  f16* rw2T = (f16*)alloc((size_t)1024 * 1024 * 2);
  f16* wxT  = (f16*)alloc((size_t)3072 * 2080 * 2);     // row stride padded 2054->2080 (16B align)
  f16* uhT  = (f16*)alloc((size_t)8 * 128 * 384 * 2);
  float* hA    = (float*)alloc(WB * RECN * 4);
  float* hB    = (float*)alloc(WB * RECN * 4);
  float* xpost = (float*)alloc(WB * 2048 * 4);
  float* ph1   = (float*)alloc(WB * 1024 * 4);
  float* plog  = (float*)alloc(WB * 1024 * 4);
  float* rh1   = (float*)alloc(WB * 1024 * 4);
  float* rlog  = (float*)alloc(WB * 1024 * 4);
  float* smp   = (float*)alloc(WB * 1024 * 4);
  float* xfull = (float*)alloc(WB * 2080 * 4);
  float* gx    = (float*)alloc(WB * 3072 * 4);
  float* gh    = (float*)alloc(WB * 3072 * 4);

  auto blks = [](long n) { return (unsigned)((n + 255) / 256); };
  // --- weight conversion: conv weights already [N][K]; dense weights transposed to [N][K] f16
  f2h_kernel<<<blks(32 * 48), 256, 0, stream>>>(enc_w[0], w1c, 32 * 48);
  f2h_kernel<<<blks(64 * 512), 256, 0, stream>>>(enc_w[1], w2c, 64 * 512);
  f2h_kernel<<<blks(128 * 1024), 256, 0, stream>>>(enc_w[2], w3c, 128 * 1024);
  f2h_kernel<<<blks(256 * 2048), 256, 0, stream>>>(enc_w[3], w4c, 256 * 2048);
  tr_f2h_kernel<<<dim3(blks(4096L * 1024), 1, 1), 256, 0, stream>>>(enc_wl, wlT, 4096, 1024, 4096);
  tr_f2h_kernel<<<dim3(blks(2048L * 1024), 1, 1), 256, 0, stream>>>(post_w1, pw1T, 2048, 1024, 2048);
  tr_f2h_kernel<<<dim3(blks(1024L * 1024), 1, 1), 256, 0, stream>>>(post_w2, pw2T, 1024, 1024, 1024);
  tr_f2h_kernel<<<dim3(blks(1024L * 1024), 1, 1), 256, 0, stream>>>(prior_w1, rw1T, 1024, 1024, 1024);
  tr_f2h_kernel<<<dim3(blks(1024L * 1024), 1, 1), 256, 0, stream>>>(prior_w2, rw2T, 1024, 1024, 1024);
  tr_f2h_kernel<<<dim3(blks(2054L * 3072), 1, 1), 256, 0, stream>>>(gru_wx, wxT, 2054, 3072, 2080);
  tr_f2h_kernel<<<dim3(blks(128L * 384), 1, 8), 256, 0, stream>>>(gru_uh, uhT, 128, 384, 128);

  // --- encoder (chunked over images to bound workspace) ---
  for (int c0 = 0; c0 < WB * WT; c0 += CH) {
    const float* obs_c = obs + (long)c0 * 3 * 64 * 64;
    conv_wmma<float><<<dim3(1, (CH * 1024) / 64), 128, 0, stream>>>(
        obs_c, w1c, enc_b[0], a1c, CH, 3, 64, 64, 32, 32, 32);
    conv_wmma<f16><<<dim3(1, (CH * 256) / 64), 128, 0, stream>>>(
        a1c, w2c, enc_b[1], a2c, CH, 32, 32, 32, 64, 16, 16);
    conv_wmma<f16><<<dim3(2, (CH * 64) / 64), 128, 0, stream>>>(
        a2c, w3c, enc_b[2], a3c, CH, 64, 16, 16, 128, 8, 8);
    conv_wmma<f16><<<dim3(4, (CH * 16) / 64), 128, 0, stream>>>(
        a3c, w4c, enc_b[3], a4c, CH, 128, 8, 8, 256, 4, 4);
    gemm_wmma<f16><<<dim3(16, CH / 64), 128, 0, stream>>>(
        a4c, 4096, 0L, wlT, 4096, 0L, enc_bl,
        embed + (long)c0 * 1024, 1024, 0L, CH, 1024, 4096, 0);
  }

  // --- sequential scan: h0 = 0; all M=32 GEMMs use the 32x128-tile kernel ---
  zero_kernel<<<blks(WB * RECN), 256, 0, stream>>>(hA, WB * RECN);
  float* h = hA; float* hn = hB;
  for (int t = 0; t < WT; ++t) {
    pack_xpost<<<blks(WB * 2048), 256, 0, stream>>>(h, embed, xpost, t);
    gemm_wmma_m32<<<dim3(8, 1), 128, 0, stream>>>(
        xpost, 2048, 0L, pw1T, 2048, 0L, post_b1, ph1, 1024, 0L, WB, 1024, 2048, 1);
    gemm_wmma_m32<<<dim3(8, 1), 128, 0, stream>>>(
        ph1, 1024, 0L, pw2T, 1024, 0L, post_b2, plog, 1024, 0L, WB, 1024, 1024, 0);
    gemm_wmma_m32<<<dim3(8, 1), 128, 0, stream>>>(
        h, 1024, 0L, rw1T, 1024, 0L, prior_b1, rh1, 1024, 0L, WB, 1024, 1024, 1);
    gemm_wmma_m32<<<dim3(8, 1), 128, 0, stream>>>(
        rh1, 1024, 0L, rw2T, 1024, 0L, prior_b2, rlog, 1024, 0L, WB, 1024, 1024, 0);
    softmax_sample<<<256, 128, 0, stream>>>(plog, rlog, h, smp, out, t);
    pack_xfull<<<blks(WB * 2080), 256, 0, stream>>>(h, smp, actions, xfull, t);
    gemm_wmma_m32<<<dim3(24, 1), 128, 0, stream>>>(
        xfull, 2080, 0L, wxT, 2080, 0L, gru_bx, gx, 3072, 0L, WB, 3072, 2054, 0);
    // block-diagonal GRU hidden term: 8 batched 32x384x128 GEMMs via blockIdx.z
    gemm_wmma_m32<<<dim3(3, 1, 8), 128, 0, stream>>>(
        h, 1024, 128L, uhT, 128, (long)384 * 128, (const float*)nullptr,
        gh, 3072, 384L, WB, 384, 128, 0);
    gru_combine<<<blks(WB * RECN), 256, 0, stream>>>(gx, gh, h, hn);
    float* tmp = h; h = hn; hn = tmp;
  }
}
